// BitLinear158_3427383902229
// MI455X (gfx1250) — 8-bit matrix operand order repaired and checked by reading, compiled, not run on hardware
//
#include <hip/hip_runtime.h>
#include <cstdint>
#include <cstddef>

typedef int v8i __attribute__((ext_vector_type(8)));

#define D_K 2048
#define NTHREADS 256
#define ELEMS_PER_THREAD (D_K / NTHREADS)

__global__ void __launch_bounds__(NTHREADS, 1)
quant_ln_kernel(const float* __restrict__ x,
                int8_t* __restrict__ xq,
                float* __restrict__ xscale)
{
    const int row  = blockIdx.x;
    const int tid  = threadIdx.x;
    const int lane = tid & 31;
    const int wave = tid >> 5;

    __shared__ float red[16];

    const float* xr = x + (size_t)row * D_K;

    float v[ELEMS_PER_THREAD];
    float sum = 0.f, sumsq = 0.f;
#pragma unroll
    for (int j = 0; j < ELEMS_PER_THREAD; ++j) {
        float f = xr[tid + j * NTHREADS];
        v[j] = f;
        sum += f;
        sumsq += f * f;
    }
#pragma unroll
    for (int o = 16; o > 0; o >>= 1) {
        sum   += __shfl_xor(sum,   o, 32);
        sumsq += __shfl_xor(sumsq, o, 32);
    }
    if (lane == 0) { red[wave] = sum; red[wave + 8] = sumsq; }
    __syncthreads();
    if (tid == 0) {
        float s = 0.f, q = 0.f;
#pragma unroll
        for (int w = 0; w < 8; ++w) { s += red[w]; q += red[w + 8]; }
        red[0] = s; red[8] = q;
    }
    __syncthreads();
    const float mu  = red[0] * (1.0f / D_K);
    const float var = fmaxf(red[8] * (1.0f / D_K) - mu * mu, 0.0f);
    const float rstd = rsqrtf(var + 1e-5f);
    __syncthreads();

    float amax = 0.f;
#pragma unroll
    for (int j = 0; j < ELEMS_PER_THREAD; ++j) {
        v[j] = (v[j] - mu) * rstd;
        amax = fmaxf(amax, fabsf(v[j]));
    }
#pragma unroll
    for (int o = 16; o > 0; o >>= 1)
        amax = fmaxf(amax, __shfl_xor(amax, o, 32));
    if (lane == 0) red[wave] = amax;
    __syncthreads();
    if (tid == 0) {
        float a = red[0];
#pragma unroll
        for (int w = 1; w < 8; ++w) a = fmaxf(a, red[w]);
        red[0] = a;
    }
    __syncthreads();
    const float clipped = fmaxf(red[0], 1e-5f);
    const float scale   = clipped * (1.0f / 127.0f);
    const float inv     = 127.0f / clipped;

    int8_t* qr = xq + (size_t)row * D_K;
#pragma unroll
    for (int j = 0; j < ELEMS_PER_THREAD; ++j) {
        float q = rintf(v[j] * inv);
        q = fminf(fmaxf(q, -127.0f), 127.0f);
        qr[tid + j * NTHREADS] = (int8_t)q;
    }
    if (tid == 0) xscale[row] = scale;
}

__global__ void __launch_bounds__(NTHREADS, 1)
w_to_i8_kernel(const int* __restrict__ wt, int8_t* __restrict__ w8, int n)
{
    int i = blockIdx.x * blockDim.x + threadIdx.x;
    if (i < n) w8[i] = (int8_t)wt[i];
}

__global__ void __launch_bounds__(NTHREADS, 1)
bitlinear_gemm_kernel(const int8_t* __restrict__ Aq,
                      const int8_t* __restrict__ W8,
                      const float*  __restrict__ xs,
                      const float*  __restrict__ wscale,
                      float* __restrict__ out,
                      int N)
{
    const int tid  = threadIdx.x;
    const int lane = tid & 31;
    const int wave = tid >> 5;
    const int half = lane >> 4;
    const int nl   = lane & 15;

    const int mBase = (blockIdx.y * 8 + wave) * 16;
    const int nBase = blockIdx.x * 64;

    const int8_t* arow = Aq + (size_t)(mBase + nl) * D_K + half * 16;
    const int8_t* brow0 = W8 + (size_t)(nBase +  0 + nl) * D_K + half * 16;
    const int8_t* brow1 = W8 + (size_t)(nBase + 16 + nl) * D_K + half * 16;
    const int8_t* brow2 = W8 + (size_t)(nBase + 32 + nl) * D_K + half * 16;
    const int8_t* brow3 = W8 + (size_t)(nBase + 48 + nl) * D_K + half * 16;

    v8i c0 = {0,0,0,0,0,0,0,0};
    v8i c1 = {0,0,0,0,0,0,0,0};
    v8i c2 = {0,0,0,0,0,0,0,0};
    v8i c3 = {0,0,0,0,0,0,0,0};

#pragma unroll 4
    for (int k0 = 0; k0 < D_K; k0 += 64) {
        v8i a;
#pragma unroll
        for (int j = 0; j < 4; ++j) {
            int2 t = *(const int2*)(arow + k0 + (j >> 1) * 32 + (j & 1) * 8);
            a[2 * j]     = t.x;
            a[2 * j + 1] = t.y;
        }

        v8i b;
        int4 t0, t1;

        t0 = *(const int4*)(brow0 + k0);
        t1 = *(const int4*)(brow0 + k0 + 32);
        b[0]=t0.x; b[1]=t0.y; b[2]=t0.z; b[3]=t0.w;
        b[4]=t1.x; b[5]=t1.y; b[6]=t1.z; b[7]=t1.w;
        c0 = __builtin_amdgcn_wmma_i32_16x16x64_iu8(true, a, true, b, c0, false, false);

        t0 = *(const int4*)(brow1 + k0);
        t1 = *(const int4*)(brow1 + k0 + 32);
        b[0]=t0.x; b[1]=t0.y; b[2]=t0.z; b[3]=t0.w;
        b[4]=t1.x; b[5]=t1.y; b[6]=t1.z; b[7]=t1.w;
        c1 = __builtin_amdgcn_wmma_i32_16x16x64_iu8(true, a, true, b, c1, false, false);

        t0 = *(const int4*)(brow2 + k0);
        t1 = *(const int4*)(brow2 + k0 + 32);
        b[0]=t0.x; b[1]=t0.y; b[2]=t0.z; b[3]=t0.w;
        b[4]=t1.x; b[5]=t1.y; b[6]=t1.z; b[7]=t1.w;
        c2 = __builtin_amdgcn_wmma_i32_16x16x64_iu8(true, a, true, b, c2, false, false);

        t0 = *(const int4*)(brow3 + k0);
        t1 = *(const int4*)(brow3 + k0 + 32);
        b[0]=t0.x; b[1]=t0.y; b[2]=t0.z; b[3]=t0.w;
        b[4]=t1.x; b[5]=t1.y; b[6]=t1.z; b[7]=t1.w;
        c3 = __builtin_amdgcn_wmma_i32_16x16x64_iu8(true, a, true, b, c3, false, false);
    }

    const float ws0 = wscale[nBase +  0 + nl];
    const float ws1 = wscale[nBase + 16 + nl];
    const float ws2 = wscale[nBase + 32 + nl];
    const float ws3 = wscale[nBase + 48 + nl];

#pragma unroll
    for (int r = 0; r < 8; ++r) {
        const int m = mBase + r + 8 * half;
        const float s = xs[m];
        float* orow = out + (size_t)m * N + nBase + nl;
        orow[ 0] = (float)c0[r] * s * ws0;
        orow[16] = (float)c1[r] * s * ws1;
        orow[32] = (float)c2[r] * s * ws2;
        orow[48] = (float)c3[r] * s * ws3;
    }
}

__global__ void __launch_bounds__(NTHREADS, 1)
out_ln_kernel(float* __restrict__ out)
{
    const int row  = blockIdx.x;
    const int tid  = threadIdx.x;
    const int lane = tid & 31;
    const int wave = tid >> 5;

    __shared__ float red[16];

    float* orow = out + (size_t)row * D_K;

    float v[ELEMS_PER_THREAD];
    float sum = 0.f, sumsq = 0.f;
#pragma unroll
    for (int j = 0; j < ELEMS_PER_THREAD; ++j) {
        float f = orow[tid + j * NTHREADS];
        v[j] = f;
        sum += f;
        sumsq += f * f;
    }
#pragma unroll
    for (int o = 16; o > 0; o >>= 1) {
        sum   += __shfl_xor(sum,   o, 32);
        sumsq += __shfl_xor(sumsq, o, 32);
    }
    if (lane == 0) { red[wave] = sum; red[wave + 8] = sumsq; }
    __syncthreads();
    if (tid == 0) {
        float s = 0.f, q = 0.f;
#pragma unroll
        for (int w = 0; w < 8; ++w) { s += red[w]; q += red[w + 8]; }
        red[0] = s; red[8] = q;
    }
    __syncthreads();
    const float mu   = red[0] * (1.0f / D_K);
    const float var  = fmaxf(red[8] * (1.0f / D_K) - mu * mu, 0.0f);
    const float rstd = rsqrtf(var + 1e-5f);

#pragma unroll
    for (int j = 0; j < ELEMS_PER_THREAD; ++j)
        orow[tid + j * NTHREADS] = (v[j] - mu) * rstd;
}

extern "C" void kernel_launch(void* const* d_in, const int* in_sizes, int n_in,
                              void* d_out, int out_size, void* d_ws, size_t ws_size,
                              hipStream_t stream)
{
    const float* x      = (const float*)d_in[0];
    const int*   wt     = (const int*)d_in[1];
    const float* wscale = (const float*)d_in[2];

    const int DOUT = in_sizes[2];
    const int K    = in_sizes[1] / DOUT;
    const int M    = in_sizes[0] / K;

    float* out = (float*)d_out;

    int8_t* xq     = (int8_t*)d_ws;
    float*  xs     = (float*)((char*)d_ws + (size_t)M * K);
    int8_t* w8     = (int8_t*)((char*)d_ws + (size_t)M * K + (size_t)M * sizeof(float));

    quant_ln_kernel<<<M, NTHREADS, 0, stream>>>(x, xq, xs);

    const int wN = DOUT * K;
    w_to_i8_kernel<<<(wN + NTHREADS - 1) / NTHREADS, NTHREADS, 0, stream>>>(wt, w8, wN);

    dim3 grid(DOUT / 64, M / 128);
    bitlinear_gemm_kernel<<<grid, NTHREADS, 0, stream>>>(xq, w8, xs, wscale, out, DOUT);

    out_ln_kernel<<<M, NTHREADS, 0, stream>>>(out);
}
